// ManualSparseAttention_79405355368909
// MI455X (gfx1250) — matrix operand order repaired and checked by reading, not compiled, not run on hardware
//
#include <hip/hip_runtime.h>
#include <hip/hip_bf16.h>
#include <cstdint>
#include <cstddef>

// ---------------------------------------------------------------------------
// MI455X (gfx1250) sliding-window attention, bf16 WMMA pipeline.
//   convert weights f32 -> bf16 transposed (hardware bf16 converts)
//   GEMM (WMMA bf16, f32 acc, register double-buffered LDS staging):
//     q, k (bf16 row-major), v (bf16 transposed per head)
//   flash-style windowed attention: one wave per 16-query tile, scores
//     computed transposed so P lands directly in the WMMA A-fragment layout;
//     wave-uniform control kept scalar so EXEC is all-1s at every WMMA
//   final GEMM: ctx(bf16) @ Wo^T + bo -> f32 d_out
// ---------------------------------------------------------------------------

typedef __attribute__((ext_vector_type(16))) __bf16       v16bf;
typedef __attribute__((ext_vector_type(8)))  float        v8f;
typedef __attribute__((ext_vector_type(4)))  unsigned int u32x4;

#define D_MODEL 1024
#define S_LEN   2048
#define BATCH   2
#define N_HEADS 16
#define D_KH    64
#define WINDOW  128

__device__ __forceinline__ unsigned short f2bf(float f) {
  return __builtin_bit_cast(unsigned short, (__bf16)f);   // native cvt, RNE
}
__device__ __forceinline__ unsigned pack2bf(float a, float b) {
  return (unsigned)f2bf(a) | ((unsigned)f2bf(b) << 16);
}
__device__ __forceinline__ v8f zero8() {
  v8f z = {0.f,0.f,0.f,0.f,0.f,0.f,0.f,0.f};
  return z;
}

union FragA {                 // 16 bf16 = 32B operand for wmma 16x16x32
  v16bf v;
  u32x4 q[2];
  unsigned u[8];
};

// ---------------------------------------------------------------------------
// Weight convert+transpose: Wt[n][k] = bf16(W[k][n]); blockIdx.y selects
// which of the four weight matrices.
// ---------------------------------------------------------------------------
__global__ __launch_bounds__(256)
void convert_weights(const float* __restrict__ w0, const float* __restrict__ w1,
                     const float* __restrict__ w2, const float* __restrict__ w3,
                     unsigned short* __restrict__ t0, unsigned short* __restrict__ t1,
                     unsigned short* __restrict__ t2, unsigned short* __restrict__ t3)
{
  const float* w; unsigned short* t;
  switch (blockIdx.y) {
    case 0:  w = w0; t = t0; break;
    case 1:  w = w1; t = t1; break;
    case 2:  w = w2; t = t2; break;
    default: w = w3; t = t3; break;
  }
  const int idx = blockIdx.x * 256 + threadIdx.x;   // over 1024*1024
  const int k = idx >> 10, n = idx & 1023;
  t[(size_t)n * D_MODEL + k] = f2bf(w[idx]);
}

// ---------------------------------------------------------------------------
// Tiled WMMA GEMM: C[4096,1024] = A[4096,1024] @ W + bias, W pre-transposed
// as bf16 Wt[n][k].  Block tile 128x128, BK=32, 256 threads (8 waves, 2Mx4N),
// each wave owns a 64x32 region = 4x2 WMMA tiles.  Global loads for tile k+1
// are issued before the WMMAs of tile k (register double buffering).
// AMODE: 0 = A f32 (convert on load), 1 = A bf16.
// OMODE: 0 = bf16 row-major, 1 = bf16 transposed-per-head [b][h][d][s],
//        2 = f32 row-major (final output).
// ---------------------------------------------------------------------------
template<int AMODE, int OMODE>
__global__ __launch_bounds__(256)
void gemm_bf16_wmma(const void* __restrict__ Aptr,
                    const unsigned short* __restrict__ Wt,
                    const float* __restrict__ bias,
                    void* __restrict__ Cptr)
{
  const int Kd = D_MODEL;
  const int N  = D_MODEL;

  // 40-half row stride: 80B -> 16 fragment rows hit disjoint bank groups
  __shared__ unsigned short As[128][40];
  __shared__ unsigned short Bs[128][40];

  const int bm   = blockIdx.x * 128;
  const int bn   = blockIdx.y * 128;
  const int tid  = threadIdx.x;
  const int lane = tid & 31;
  const int wave = tid >> 5;
  const int wm   = (wave >> 2) * 64;   // 0 / 64
  const int wn   = (wave & 3) * 32;    // 0 / 32 / 64 / 96
  const int r16  = lane & 15;
  const int hi   = lane >> 4;          // half-wave select

  int rr[4], cc4[4];
  #pragma unroll
  for (int i = 0; i < 4; ++i) {
    const int flat = tid + i * 256;    // 1024 16B groups per tile
    rr[i]  = flat >> 3;
    cc4[i] = (flat & 7) << 2;
  }

  v8f acc[4][2];
  #pragma unroll
  for (int mt = 0; mt < 4; ++mt)
    #pragma unroll
    for (int nt = 0; nt < 2; ++nt) acc[mt][nt] = zero8();

  float4  af32[4];
  ushort4 a16[4];
  ushort4 b16[4];

  // ---- prologue: load tile kb=0 into registers, commit to LDS ----
  if (AMODE == 0) {
    const float* A = (const float*)Aptr;
    #pragma unroll
    for (int i = 0; i < 4; ++i)
      af32[i] = *(const float4*)(A + (size_t)(bm + rr[i]) * Kd + cc4[i]);
  } else {
    const unsigned short* A = (const unsigned short*)Aptr;
    #pragma unroll
    for (int i = 0; i < 4; ++i)
      a16[i] = *(const ushort4*)(A + (size_t)(bm + rr[i]) * Kd + cc4[i]);
  }
  #pragma unroll
  for (int i = 0; i < 4; ++i)
    b16[i] = *(const ushort4*)(Wt + (size_t)(bn + rr[i]) * Kd + cc4[i]);

  #pragma unroll
  for (int i = 0; i < 4; ++i) {
    if (AMODE == 0) {
      ushort4 p;
      p.x = f2bf(af32[i].x); p.y = f2bf(af32[i].y);
      p.z = f2bf(af32[i].z); p.w = f2bf(af32[i].w);
      *(ushort4*)&As[rr[i]][cc4[i]] = p;
    } else {
      *(ushort4*)&As[rr[i]][cc4[i]] = a16[i];
    }
    *(ushort4*)&Bs[rr[i]][cc4[i]] = b16[i];
  }
  __syncthreads();

  for (int kb = 0; kb < Kd; kb += 32) {
    const int nkb = kb + 32;

    // ---- issue next tile's global loads (latency hidden under WMMAs) ----
    if (nkb < Kd) {
      if (AMODE == 0) {
        const float* A = (const float*)Aptr;
        #pragma unroll
        for (int i = 0; i < 4; ++i) {
          const float* p = A + (size_t)(bm + rr[i]) * Kd + nkb + cc4[i];
          af32[i] = *(const float4*)p;
          __builtin_prefetch(p + 32, 0, 1);   // one more tile ahead -> L2
        }
      } else {
        const unsigned short* A = (const unsigned short*)Aptr;
        #pragma unroll
        for (int i = 0; i < 4; ++i) {
          const unsigned short* p = A + (size_t)(bm + rr[i]) * Kd + nkb + cc4[i];
          a16[i] = *(const ushort4*)p;
          __builtin_prefetch(p + 32, 0, 1);
        }
      }
      #pragma unroll
      for (int i = 0; i < 4; ++i)
        b16[i] = *(const ushort4*)(Wt + (size_t)(bn + rr[i]) * Kd + nkb + cc4[i]);
    }

    // ---- fragments from LDS ----
    FragA af[4], bfr[2];
    #pragma unroll
    for (int mt = 0; mt < 4; ++mt) {
      const unsigned short* p = &As[wm + mt * 16 + r16][hi * 8];
      af[mt].q[0] = *(const u32x4*)p;          // K = kb8 .. kb8+7
      af[mt].q[1] = *(const u32x4*)(p + 16);   // K = 16+kb8 .. 16+kb8+7
    }
    #pragma unroll
    for (int nt = 0; nt < 2; ++nt) {
      const unsigned short* p = &Bs[wn + nt * 16 + r16][hi * 8];
      bfr[nt].q[0] = *(const u32x4*)p;         // K = hi*8 .. +7
      bfr[nt].q[1] = *(const u32x4*)(p + 16);  // K = 16+hi*8 .. +7
    }
    #pragma unroll
    for (int mt = 0; mt < 4; ++mt)
      #pragma unroll
      for (int nt = 0; nt < 2; ++nt)
        acc[mt][nt] = __builtin_amdgcn_wmma_f32_16x16x32_bf16(
            false, af[mt].v, false, bfr[nt].v, (short)0, acc[mt][nt],
            false, false);

    // ---- commit next tile to LDS ----
    if (nkb < Kd) {
      __syncthreads();
      #pragma unroll
      for (int i = 0; i < 4; ++i) {
        if (AMODE == 0) {
          ushort4 p;
          p.x = f2bf(af32[i].x); p.y = f2bf(af32[i].y);
          p.z = f2bf(af32[i].z); p.w = f2bf(af32[i].w);
          *(ushort4*)&As[rr[i]][cc4[i]] = p;
        } else {
          *(ushort4*)&As[rr[i]][cc4[i]] = a16[i];
        }
        *(ushort4*)&Bs[rr[i]][cc4[i]] = b16[i];
      }
      __syncthreads();
    }
  }

  // ---- store: C layout lane = N column, VGPR r = M row (+8 for upper half) --
  #pragma unroll
  for (int nt = 0; nt < 2; ++nt) {
    const int gc = bn + wn + nt * 16 + r16;
    const float bv = bias[gc];
    #pragma unroll
    for (int mt = 0; mt < 4; ++mt) {
      #pragma unroll
      for (int vr = 0; vr < 8; ++vr) {
        const int gr = bm + wm + mt * 16 + vr + hi * 8;
        const float val = acc[mt][nt][vr] + bv;
        if (OMODE == 0) {
          ((unsigned short*)Cptr)[(size_t)gr * N + gc] = f2bf(val);
        } else if (OMODE == 2) {
          ((float*)Cptr)[(size_t)gr * N + gc] = val;
        } else { // transposed-per-head v^T: [b][h][d][s]
          const int b = gr >> 11, s = gr & (S_LEN - 1);
          const int h = gc >> 6,  d = gc & (D_KH - 1);
          ((unsigned short*)Cptr)[(((size_t)(b * N_HEADS + h)) * D_KH + d) * S_LEN + s]
              = f2bf(val);
        }
      }
    }
  }
}

// ---------------------------------------------------------------------------
// Sliding-window attention, flash-style.  One wave per 16-query tile.
// Scores computed transposed: S^T[key][query] = k_tile @ q^T, so the C layout
// (lane = query, vgpr = key) matches the A-fragment layout needed for P@V.
// qs is forced scalar (readfirstlane) so all tile-validity branches are
// s_cbranch and EXEC stays all-1s at every WMMA.
// ---------------------------------------------------------------------------
__global__ __launch_bounds__(256)
void sliding_attn_wmma(const unsigned short* __restrict__ qb,
                       const unsigned short* __restrict__ kbuf,
                       const unsigned short* __restrict__ vtb,
                       unsigned short* __restrict__ ctx)
{
  const int lane = threadIdx.x & 31;
  const int wave = threadIdx.x >> 5;
  const int h  = blockIdx.y;
  const int b  = blockIdx.z;
  const int qs =
      __builtin_amdgcn_readfirstlane((blockIdx.x * 8 + wave) * 16); // scalar
  const int col = lane & 15;
  const int hi  = lane >> 4;

  // q B-fragments (reused for every key tile): lane = query col, table-order d
  FragA qf[2];
  {
    const unsigned short* qp =
        qb + ((size_t)(b * S_LEN) + qs + col) * D_MODEL + h * D_KH;
    #pragma unroll
    for (int kc = 0; kc < 2; ++kc) {
      qf[kc].q[0] = *(const u32x4*)(qp + kc * 32 + hi * 8);
      qf[kc].q[1] = *(const u32x4*)(qp + kc * 32 + hi * 8 + 16);
    }
  }

  float m_i = -1e30f, l_i = 0.0f;
  v8f o[4];
  #pragma unroll
  for (int nt = 0; nt < 4; ++nt) o[nt] = zero8();

  const float SC = 0.125f * 1.44269504088896340736f; // 1/sqrt(64) * log2(e)
  const int q_abs = qs + col;

  for (int c = 0; c < 9; ++c) {                 // 9 chunks of 32 keys
    const int kb0 = qs - WINDOW + c * 32;       // scalar
    const bool v0 = (kb0 >= 0) && (kb0 < S_LEN);
    const bool v1 = (kb0 + 16 >= 0) && (kb0 + 16 < S_LEN);
    if (!v0 && !v1) continue;                   // scalar branch

    v8f s0 = zero8(), s1 = zero8();
    if (v0) {
      const unsigned short* kp =
          kbuf + ((size_t)(b * S_LEN) + kb0 + col) * D_MODEL + h * D_KH;
      FragA kf0, kf1;
      kf0.q[0] = *(const u32x4*)(kp + hi * 8);
      kf0.q[1] = *(const u32x4*)(kp + hi * 8 + 16);
      kf1.q[0] = *(const u32x4*)(kp + 32 + hi * 8);
      kf1.q[1] = *(const u32x4*)(kp + 32 + hi * 8 + 16);
      s0 = __builtin_amdgcn_wmma_f32_16x16x32_bf16(
          false, kf0.v, false, qf[0].v, (short)0, s0, false, false);
      s0 = __builtin_amdgcn_wmma_f32_16x16x32_bf16(
          false, kf1.v, false, qf[1].v, (short)0, s0, false, false);
    }
    if (v1) {
      const unsigned short* kp =
          kbuf + ((size_t)(b * S_LEN) + kb0 + 16 + col) * D_MODEL + h * D_KH;
      FragA kf0, kf1;
      kf0.q[0] = *(const u32x4*)(kp + hi * 8);
      kf0.q[1] = *(const u32x4*)(kp + hi * 8 + 16);
      kf1.q[0] = *(const u32x4*)(kp + 32 + hi * 8);
      kf1.q[1] = *(const u32x4*)(kp + 32 + hi * 8 + 16);
      s1 = __builtin_amdgcn_wmma_f32_16x16x32_bf16(
          false, kf0.v, false, qf[0].v, (short)0, s1, false, false);
      s1 = __builtin_amdgcn_wmma_f32_16x16x32_bf16(
          false, kf1.v, false, qf[1].v, (short)0, s1, false, false);
    }

    // ---- V fragments (masked) loaded up front: no VALU between PV WMMAs ----
    const int  k0 = v0 ? (kb0 + hi * 8) : 0;
    const int  k1 = v1 ? (kb0 + 16 + hi * 8) : 0;
    u32x4 z4 = {0u, 0u, 0u, 0u};
    FragA vf[4];
    #pragma unroll
    for (int nt = 0; nt < 4; ++nt) {
      const unsigned short* vp =
          vtb + (((size_t)(b * N_HEADS + h)) * D_KH + nt * 16 + col) * S_LEN;
      vf[nt].q[0] = *(const u32x4*)(vp + k0);
      vf[nt].q[1] = *(const u32x4*)(vp + k1);
      if (!v0) vf[nt].q[0] = z4;
      if (!v1) vf[nt].q[1] = z4;
    }

    // mask + scale (exp2 domain) + per-chunk row max
    float p0[8], p1[8];
    float cm = -1e30f;
    #pragma unroll
    for (int vr = 0; vr < 8; ++vr) {
      const int key0 = kb0 + vr + hi * 8;
      const int rel0 = key0 - q_abs;
      const bool ok0 = v0 && (rel0 >= -WINDOW) && (rel0 <= WINDOW);
      const float sh0 = ok0 ? s0[vr] * SC : -1e30f;
      p0[vr] = sh0; cm = fmaxf(cm, sh0);
      const int rel1 = rel0 + 16;
      const bool ok1 = v1 && (rel1 >= -WINDOW) && (rel1 <= WINDOW);
      const float sh1 = ok1 ? s1[vr] * SC : -1e30f;
      p1[vr] = sh1; cm = fmaxf(cm, sh1);
    }
    cm = fmaxf(cm, __shfl_xor(cm, 16, 32));     // combine key halves

    const float m_new   = fmaxf(m_i, cm);
    const float rescale = exp2f(m_i - m_new);

    // P = exp2(sHat - m_new), packed straight into the A-fragment layout
    FragA pf;
    float rsum = 0.0f;
    #pragma unroll
    for (int j = 0; j < 4; ++j) {
      const float e0a = (p0[2*j]   > -1e29f) ? exp2f(p0[2*j]   - m_new) : 0.f;
      const float e0b = (p0[2*j+1] > -1e29f) ? exp2f(p0[2*j+1] - m_new) : 0.f;
      const float e1a = (p1[2*j]   > -1e29f) ? exp2f(p1[2*j]   - m_new) : 0.f;
      const float e1b = (p1[2*j+1] > -1e29f) ? exp2f(p1[2*j+1] - m_new) : 0.f;
      rsum += e0a + e0b + e1a + e1b;
      pf.u[j]     = pack2bf(e0a, e0b);          // keys kb8+2j, kb8+2j+1
      pf.u[4 + j] = pack2bf(e1a, e1b);          // keys 16+kb8+2j, +2j+1
    }
    rsum += __shfl_xor(rsum, 16, 32);
    l_i = l_i * rescale + rsum;
    m_i = m_new;

    // rescale O rows (rows are VGPRs here -> broadcast per-query scale)
    float rs[8];
    #pragma unroll
    for (int vr = 0; vr < 8; ++vr) rs[vr] = __shfl(rescale, vr + hi * 8, 32);
    #pragma unroll
    for (int nt = 0; nt < 4; ++nt)
      #pragma unroll
      for (int vr = 0; vr < 8; ++vr) o[nt][vr] *= rs[vr];

    // P @ V over 32 keys, back-to-back WMMAs on independent accumulators
    #pragma unroll
    for (int nt = 0; nt < 4; ++nt)
      o[nt] = __builtin_amdgcn_wmma_f32_16x16x32_bf16(
          false, pf.v, false, vf[nt].v, (short)0, o[nt], false, false);
  }

  // normalize and store context (bf16 row-major [b][s][h*64+d])
  const float linv = 1.0f / l_i;
  float rl[8];
  #pragma unroll
  for (int vr = 0; vr < 8; ++vr) rl[vr] = __shfl(linv, vr + hi * 8, 32);
  #pragma unroll
  for (int nt = 0; nt < 4; ++nt) {
    #pragma unroll
    for (int vr = 0; vr < 8; ++vr) {
      const int row = qs + vr + hi * 8;
      const int cc  = h * D_KH + nt * 16 + col;
      ctx[((size_t)(b * S_LEN) + row) * D_MODEL + cc] = f2bf(o[nt][vr] * rl[vr]);
    }
  }
}

// ---------------------------------------------------------------------------
// Host-side orchestration (all on `stream`, graph-capture safe).
// ---------------------------------------------------------------------------
extern "C" void kernel_launch(void* const* d_in, const int* in_sizes, int n_in,
                              void* d_out, int out_size, void* d_ws, size_t ws_size,
                              hipStream_t stream)
{
  const float* Q  = (const float*)d_in[0];
  const float* K  = (const float*)d_in[1];
  const float* V  = (const float*)d_in[2];
  const float* Wq = (const float*)d_in[3];
  const float* bq = (const float*)d_in[4];
  const float* Wk = (const float*)d_in[5];
  const float* bk = (const float*)d_in[6];
  const float* Wv = (const float*)d_in[7];
  const float* bv = (const float*)d_in[8];
  const float* Wo = (const float*)d_in[9];
  const float* bo = (const float*)d_in[10];

  char* ws = (char*)d_ws;
  const size_t WSZ = (size_t)D_MODEL * D_MODEL * sizeof(unsigned short);          // 2 MB
  const size_t ASZ = (size_t)BATCH * S_LEN * D_MODEL * sizeof(unsigned short);    // 8 MB
  unsigned short* wq_t = (unsigned short*)(ws);
  unsigned short* wk_t = (unsigned short*)(ws + WSZ);
  unsigned short* wv_t = (unsigned short*)(ws + 2 * WSZ);
  unsigned short* wo_t = (unsigned short*)(ws + 3 * WSZ);
  unsigned short* qbuf = (unsigned short*)(ws + 4 * WSZ);
  unsigned short* kbuf = (unsigned short*)(ws + 4 * WSZ + ASZ);
  unsigned short* vtb  = (unsigned short*)(ws + 4 * WSZ + 2 * ASZ);
  unsigned short* ctx  = (unsigned short*)(ws + 4 * WSZ + 3 * ASZ);
  (void)in_sizes; (void)n_in; (void)out_size; (void)ws_size;

  // 1) weights -> bf16, transposed
  convert_weights<<<dim3((D_MODEL * D_MODEL) / 256, 4), 256, 0, stream>>>(
      Wq, Wk, Wv, Wo, wq_t, wk_t, wv_t, wo_t);

  // 2) projections (M = B*S = 4096 rows)
  dim3 gg((BATCH * S_LEN) / 128, D_MODEL / 128);
  gemm_bf16_wmma<0, 0><<<gg, 256, 0, stream>>>(Q, wq_t, bq, qbuf);
  gemm_bf16_wmma<0, 0><<<gg, 256, 0, stream>>>(K, wk_t, bk, kbuf);
  gemm_bf16_wmma<0, 1><<<gg, 256, 0, stream>>>(V, wv_t, bv, vtb);

  // 3) windowed attention (8 query tiles per 256-thread block)
  sliding_attn_wmma<<<dim3(S_LEN / 128, N_HEADS, BATCH), 256, 0, stream>>>(
      qbuf, kbuf, vtb, ctx);

  // 4) output projection -> f32
  gemm_bf16_wmma<1, 2><<<gg, 256, 0, stream>>>(ctx, wo_t, bo, d_out);
}
